// RoPEMultiHeadAttention_13615046328483
// MI455X (gfx1250) — compile-verified, matrix operand order checked by reading, not run on hardware
//
#include <hip/hip_runtime.h>
#include <hip/hip_bf16.h>
#include <math.h>

// ---------- types matching the gfx1250 WMMA builtin signatures ----------
typedef __bf16 bf16;
typedef __attribute__((ext_vector_type(16))) __bf16 v16bf;
typedef __attribute__((ext_vector_type(8)))  __bf16 v8bf;
typedef __attribute__((ext_vector_type(8)))  float  v8f;
typedef __attribute__((ext_vector_type(4)))  unsigned v4u;
typedef __attribute__((ext_vector_type(8)))  unsigned v8u;

union ABFrag {          // 16x32 (A) or 32x16 (B) bf16 fragment: 8 VGPRs
    v16bf v;
    v8bf  h[2];         // two contiguous 16-byte chunks per lane
};

#define D_MODEL 1024
#define HEADS   16
#define DKH     64
#define SEQ     2048
#define BATCH   2
#define MROWS   (BATCH * SEQ)   // 4096

static __device__ __forceinline__ v8f wmma_bf16(const ABFrag& a, const ABFrag& b, v8f c) {
    return __builtin_amdgcn_wmma_f32_16x16x32_bf16(
        /*neg_a=*/false, a.v, /*neg_b=*/false, b.v,
        /*c_mod=*/(short)0, c, /*reuse_a=*/false, /*reuse_b=*/false);
}

// 32-bit hardware LDS byte address of a generic pointer into __shared__
static __device__ __forceinline__ unsigned lds_addr_u32(const void* p) {
    const __attribute__((address_space(3))) char* lp =
        (const __attribute__((address_space(3))) char*)p;
    return (unsigned)(unsigned long long)lp;
}

// ---------------- fp32 -> bf16 convert ----------------
__global__ __launch_bounds__(256) void cvt_f32_bf16(const float* __restrict__ src,
                                                    bf16* __restrict__ dst, int n) {
    int i = blockIdx.x * 256 + threadIdx.x;
    if (i < n) dst[i] = (bf16)src[i];
}

// ---------------- bf16 WMMA GEMM: C[M][N] = A[M][K] * B[N][K]^T ----------------
__global__ __launch_bounds__(256)
void gemm_bf16_wmma(const bf16* __restrict__ A, const bf16* __restrict__ B,
                    float* __restrict__ Cf, bf16* __restrict__ Cb,
                    int M, int N, int K)
{
    const int lane  = threadIdx.x & 31;
    const int wave  = threadIdx.x >> 5;
    const int waveM = wave >> 2;                 // 0..1
    const int waveN = wave & 3;                  // 0..3
    const int hi16  = lane >> 4;
    const int l16   = lane & 15;
    const int kofs  = hi16 * 8;

    const int rowBase = blockIdx.y * 64 + waveM * 32;
    const int colBase = blockIdx.x * 128 + waveN * 32;

    v8f acc[2][2];
#pragma unroll
    for (int i = 0; i < 2; ++i)
#pragma unroll
        for (int j = 0; j < 2; ++j) acc[i][j] = {};

    for (int k0 = 0; k0 < K; k0 += 32) {
        ABFrag a[2], b[2];
#pragma unroll
        for (int i = 0; i < 2; ++i) {
            const bf16* ap = A + (size_t)(rowBase + i * 16 + l16) * K + k0;
            a[i].h[0] = *(const v8bf*)(ap + kofs);
            a[i].h[1] = *(const v8bf*)(ap + 16 + kofs);
        }
#pragma unroll
        for (int j = 0; j < 2; ++j) {
            const bf16* bp = B + (size_t)(colBase + j * 16 + l16) * K + k0;
            b[j].h[0] = *(const v8bf*)(bp + kofs);
            b[j].h[1] = *(const v8bf*)(bp + 16 + kofs);
        }
        __builtin_prefetch(A + (size_t)(rowBase + l16) * K + k0 + 32, 0, 1);
        __builtin_prefetch(B + (size_t)(colBase + l16) * K + k0 + 32, 0, 1);
#pragma unroll
        for (int i = 0; i < 2; ++i)
#pragma unroll
            for (int j = 0; j < 2; ++j)
                acc[i][j] = wmma_bf16(a[i], b[j], acc[i][j]);
    }

#pragma unroll
    for (int i = 0; i < 2; ++i)
#pragma unroll
        for (int j = 0; j < 2; ++j)
#pragma unroll
            for (int v = 0; v < 8; ++v) {
                int r = rowBase + i * 16 + v + hi16 * 8;
                int c = colBase + j * 16 + l16;
                float val = acc[i][j][v];
                if (Cf) Cf[(size_t)r * N + c] = val;
                else    Cb[(size_t)r * N + c] = (bf16)val;
            }
}

// ---------------- RoPE (reference variant) ----------------
__global__ __launch_bounds__(256)
void rope_kernel(const bf16* __restrict__ Qi, const bf16* __restrict__ Ki,
                 bf16* __restrict__ Qo, bf16* __restrict__ Ko, int n)
{
    int idx = blockIdx.x * 256 + threadIdx.x;
    if (idx >= n) return;
    int d = idx & (DKH - 1);
    int s = (idx / D_MODEL) & (SEQ - 1);
    int i = d & 31;
    float inv_freq = __powf(10000.0f, -(float)i * (1.0f / 32.0f));
    float ang = (float)s * inv_freq;
    float c, sn;
    __sincosf(ang, &sn, &c);
    int base = idx - d;
    int rot_idx = (d < 32) ? (base + 2 * d + 1) : (base + 2 * (d - 32));
    float rsign = (d < 32) ? -1.0f : 1.0f;

    float qv = (float)Qi[idx], qr = rsign * (float)Qi[rot_idx];
    float kv = (float)Ki[idx], kr = rsign * (float)Ki[rot_idx];
    Qo[idx] = (bf16)(qv * c + qr * sn);
    Ko[idx] = (bf16)(kv * c + kr * sn);
}

// ---------------- V transpose: (b,s,h,d) -> Vt[(b*H+h)*DKH + d][SEQ] ----------------
__global__ __launch_bounds__(256)
void transpose_v(const bf16* __restrict__ V, bf16* __restrict__ Vt, int n)
{
    int idx = blockIdx.x * 256 + threadIdx.x;
    if (idx >= n) return;
    int d  = idx & (DKH - 1);
    int h  = (idx >> 6) & (HEADS - 1);
    int s  = (idx / D_MODEL) & (SEQ - 1);
    int b  = idx / (SEQ * D_MODEL);
    Vt[((size_t)(b * HEADS + h) * DKH + d) * SEQ + s] = V[idx];
}

// ---------------- flash attention with async-LDS (K) + TDM (V) staging ----------------
// One wave per (b,h, 16-query tile). Per 32-key block:
//   K tile (32x64 bf16, 4KB)  : global_load_async_to_lds_b128  (ASYNCcnt)
//   V tile (64x32 bf16, 4KB)  : tensor_load_to_lds via TDM D#  (TENSORcnt)
//   S = Q*K^T (4 WMMA), online softmax, P staged via LDS, O += P*V (4 WMMA)
__global__ __launch_bounds__(256)
void flash_attn_wmma(const bf16* __restrict__ Q, const bf16* __restrict__ Km,
                     const bf16* __restrict__ Vt, bf16* __restrict__ O, float scale)
{
    __shared__ __align__(16) bf16 sK[8][32 * DKH];   // per-wave K tile, row-major 32x64
    __shared__ __align__(16) bf16 sV[8][DKH * 32];   // per-wave V tile, row-major 64x32
    __shared__ __align__(16) bf16 sP[8][16 * 32];    // per-wave P staging

    const int lane = threadIdx.x & 31;
    const int wave = threadIdx.x >> 5;
    const int gw   = blockIdx.x * 8 + wave;          // 0..4095
    const int bh   = gw >> 7;                        // 0..31
    const int qt   = gw & 127;
    const int b    = bh >> 4;
    const int h    = bh & 15;
    const int hi16 = lane >> 4;
    const int l16  = lane & 15;
    const int kofs = hi16 * 8;

    bf16* kbuf = &sK[wave][0];
    bf16* vbuf = &sV[wave][0];
    bf16* pbuf = &sP[wave][0];
    const unsigned kbuf_lds = lds_addr_u32(kbuf);
    const unsigned vbuf_lds = lds_addr_u32(vbuf);

    // Q fragments for k-dim chunks [0,32) and [32,64)
    ABFrag aQ[2];
    const bf16* qp = Q + (size_t)(b * SEQ + qt * 16 + l16) * D_MODEL + h * DKH;
    aQ[0].h[0] = *(const v8bf*)(qp + kofs);
    aQ[0].h[1] = *(const v8bf*)(qp + 16 + kofs);
    aQ[1].h[0] = *(const v8bf*)(qp + 32 + kofs);
    aQ[1].h[1] = *(const v8bf*)(qp + 48 + kofs);

    v8f o[4];
#pragma unroll
    for (int t = 0; t < 4; ++t) o[t] = {};
    float mrow[8], lrow[8];
#pragma unroll
    for (int v = 0; v < 8; ++v) { mrow[v] = -3.0e38f; lrow[v] = 0.0f; }

    for (int kb = 0; kb < SEQ; kb += 32) {
        // make sure prior-iteration DS reads of kbuf/vbuf retired before overwriting
        asm volatile("s_wait_dscnt 0x0" ::: "memory");

        // ---- async copy K tile: lane = key row, 128B/row in 8x b128 chunks ----
        {
            const bf16* gk = Km + (size_t)(b * SEQ + kb + lane) * D_MODEL + h * DKH;
            unsigned lk = kbuf_lds + (unsigned)lane * (DKH * 2);
#pragma unroll
            for (int i = 0; i < 8; ++i) {
                asm volatile("global_load_async_to_lds_b128 %0, %1, off offset:%2"
                             :: "v"(lk), "v"(gk), "i"(i * 16) : "memory");
            }
        }

        // ---- TDM: V tile (64 rows x 32 cols bf16) -> vbuf ----
        {
            unsigned long long gv =
                (unsigned long long)(Vt + (size_t)bh * DKH * SEQ + kb);
            v4u g0;
            g0[0] = 1u;                                   // count=1, user descriptor
            g0[1] = vbuf_lds;                             // lds_addr
            g0[2] = (unsigned)gv;                         // global_addr[31:0]
            g0[3] = (unsigned)(gv >> 32) | 0x80000000u;   // global_addr[56:32] | type=2
            v8u g1;
            g1[0] = 0x00010000u;                          // wg_mask=0, data_size=1 (2B)
            g1[1] = ((unsigned)SEQ & 0xffffu) << 16;      // tensor_dim0[15:0] = 2048
            g1[2] = ((unsigned)(BATCH * HEADS * DKH) & 0xffffu) << 16; // tensor_dim1 lo
            g1[3] = 32u << 16;                            // tile_dim0 = 32
            g1[4] = 64u;                                  // tile_dim1 = 64, tile_dim2 = 0
            g1[5] = (unsigned)SEQ;                        // tensor_dim0_stride = 2048
            g1[6] = 0u;
            g1[7] = 0u;
            asm volatile("tensor_load_to_lds %0, %1" :: "s"(g0), "s"(g1) : "memory");
        }

        asm volatile("s_wait_asynccnt 0x0" ::: "memory");
        __builtin_amdgcn_s_wait_tensorcnt(0);

        // ---- S = Q * K^T, B-fragments from LDS K tile ----
        ABFrag bK0[2], bK1[2];
#pragma unroll
        for (int j = 0; j < 2; ++j) {
            const bf16* kp = kbuf + (j * 16 + l16) * DKH;
            bK0[j].h[0] = *(const v8bf*)(kp + kofs);
            bK0[j].h[1] = *(const v8bf*)(kp + 16 + kofs);
            bK1[j].h[0] = *(const v8bf*)(kp + 32 + kofs);
            bK1[j].h[1] = *(const v8bf*)(kp + 48 + kofs);
        }
        v8f s0 = {}, s1 = {};
        s0 = wmma_bf16(aQ[0], bK0[0], s0);
        s0 = wmma_bf16(aQ[1], bK1[0], s0);
        s1 = wmma_bf16(aQ[0], bK0[1], s1);
        s1 = wmma_bf16(aQ[1], bK1[1], s1);

        // ---- online softmax; row stats per 16-lane half via shfl_xor ----
        float alpha[8];
#pragma unroll
        for (int v = 0; v < 8; ++v) {
            float x0 = s0[v] * scale, x1 = s1[v] * scale;
            float mx = fmaxf(x0, x1);
#pragma unroll
            for (int msk = 1; msk < 16; msk <<= 1) mx = fmaxf(mx, __shfl_xor(mx, msk, 32));
            float mn = fmaxf(mrow[v], mx);
            alpha[v] = __expf(mrow[v] - mn);
            mrow[v] = mn;
            float p0 = __expf(x0 - mn), p1 = __expf(x1 - mn);
            float rs = p0 + p1;
#pragma unroll
            for (int msk = 1; msk < 16; msk <<= 1) rs += __shfl_xor(rs, msk, 32);
            lrow[v] = lrow[v] * alpha[v] + rs;
            int m = v + hi16 * 8;
            pbuf[m * 32 + l16]      = (bf16)p0;
            pbuf[m * 32 + 16 + l16] = (bf16)p1;
        }

        // ---- reload P as A-fragment ----
        ABFrag aP;
        aP.h[0] = *(const v8bf*)(pbuf + l16 * 32 + kofs);
        aP.h[1] = *(const v8bf*)(pbuf + l16 * 32 + 16 + kofs);

        // ---- rescale O and accumulate O += P * V (B-fragments from LDS V tile) ----
#pragma unroll
        for (int t = 0; t < 4; ++t)
#pragma unroll
            for (int v = 0; v < 8; ++v) o[t][v] *= alpha[v];

#pragma unroll
        for (int t = 0; t < 4; ++t) {
            const bf16* vp = vbuf + (t * 16 + l16) * 32;
            ABFrag bV;
            bV.h[0] = *(const v8bf*)(vp + kofs);
            bV.h[1] = *(const v8bf*)(vp + 16 + kofs);
            o[t] = wmma_bf16(aP, bV, o[t]);
        }
    }

    // ---- normalize and store (row-major MROWS x D_MODEL, bf16) ----
#pragma unroll
    for (int t = 0; t < 4; ++t)
#pragma unroll
        for (int v = 0; v < 8; ++v) {
            int r = b * SEQ + qt * 16 + v + hi16 * 8;
            int c = h * DKH + t * 16 + l16;
            O[(size_t)r * D_MODEL + c] = (bf16)(o[t][v] / lrow[v]);
        }
}

// ---------------- host-side launch ----------------
extern "C" void kernel_launch(void* const* d_in, const int* in_sizes, int n_in,
                              void* d_out, int out_size, void* d_ws, size_t ws_size,
                              hipStream_t stream)
{
    (void)in_sizes; (void)n_in; (void)out_size; (void)ws_size;
    const float* x  = (const float*)d_in[0];
    const float* Wq = (const float*)d_in[1];
    const float* Wk = (const float*)d_in[2];
    const float* Wv = (const float*)d_in[3];
    const float* Wo = (const float*)d_in[4];

    const size_t NX = (size_t)MROWS * D_MODEL;      // 4M
    const size_t NW = (size_t)D_MODEL * D_MODEL;    // 1M

    char* ws = (char*)d_ws;
    size_t off = 0;
    bf16* Xb  = (bf16*)(ws + off); off += NX * 2;
    bf16* Wqb = (bf16*)(ws + off); off += NW * 2;
    bf16* Wkb = (bf16*)(ws + off); off += NW * 2;
    bf16* Wvb = (bf16*)(ws + off); off += NW * 2;
    bf16* Wob = (bf16*)(ws + off); off += NW * 2;
    bf16* Qb0 = (bf16*)(ws + off); off += NX * 2;
    bf16* Kb0 = (bf16*)(ws + off); off += NX * 2;
    bf16* Vb0 = (bf16*)(ws + off); off += NX * 2;
    bf16* Qr  = (bf16*)(ws + off); off += NX * 2;
    bf16* Kr  = (bf16*)(ws + off); off += NX * 2;
    bf16* Vt  = (bf16*)(ws + off); off += NX * 2;
    bf16* Ab  = (bf16*)(ws + off); off += NX * 2;

    // 1) fp32 -> bf16
    cvt_f32_bf16<<<(int)(NX / 256), 256, 0, stream>>>(x,  Xb,  (int)NX);
    cvt_f32_bf16<<<(int)(NW / 256), 256, 0, stream>>>(Wq, Wqb, (int)NW);
    cvt_f32_bf16<<<(int)(NW / 256), 256, 0, stream>>>(Wk, Wkb, (int)NW);
    cvt_f32_bf16<<<(int)(NW / 256), 256, 0, stream>>>(Wv, Wvb, (int)NW);
    cvt_f32_bf16<<<(int)(NW / 256), 256, 0, stream>>>(Wo, Wob, (int)NW);

    // 2) projections: Q/K/V = X @ W^T
    dim3 ggrid(D_MODEL / 128, MROWS / 64);
    gemm_bf16_wmma<<<ggrid, 256, 0, stream>>>(Xb, Wqb, nullptr, Qb0, MROWS, D_MODEL, D_MODEL);
    gemm_bf16_wmma<<<ggrid, 256, 0, stream>>>(Xb, Wkb, nullptr, Kb0, MROWS, D_MODEL, D_MODEL);
    gemm_bf16_wmma<<<ggrid, 256, 0, stream>>>(Xb, Wvb, nullptr, Vb0, MROWS, D_MODEL, D_MODEL);

    // 3) RoPE on Q,K ; 4) transpose V
    rope_kernel<<<(int)(NX / 256), 256, 0, stream>>>(Qb0, Kb0, Qr, Kr, (int)NX);
    transpose_v<<<(int)(NX / 256), 256, 0, stream>>>(Vb0, Vt, (int)NX);

    // 5) flash attention (4096 wave-tasks, 8 waves per block)
    flash_attn_wmma<<<512, 256, 0, stream>>>(Qr, Kr, Vt, Ab, 0.125f);

    // 6) output projection -> d_out (fp32)
    gemm_bf16_wmma<<<ggrid, 256, 0, stream>>>(Ab, Wob, (float*)d_out, nullptr, MROWS, D_MODEL, D_MODEL);
}
